// DynamicConvBranch_25451976196560
// MI455X (gfx1250) — matrix operand order repaired and checked by reading, not compiled, not run on hardware
//
#include <hip/hip_runtime.h>

typedef __attribute__((ext_vector_type(16))) _Float16 v16h;
typedef __attribute__((ext_vector_type(8)))  float    v8f;
typedef __attribute__((ext_vector_type(4)))  float    f4;   // ext-vector float4 (NT-capable)

#define B_   16
#define C_   32
#define H_   384
#define W_   384
#define NK_  32
#define HID_ 128
#define KK_  288   // NK * 3 * 3

// ---------------------------------------------------------------------------
// Kernel 1: global average pool.  One block per (b,c) plane, b128 streaming
// loads with non-temporal hint (no reuse inside this pass; x > L2 anyway),
// LDS tree reduction.  302 MB read -> ~13 us at 23.3 TB/s.
// ---------------------------------------------------------------------------
__global__ void pool_kernel(const float* __restrict__ x, float* __restrict__ pooled) {
    __shared__ float red[256];
    const int plane = blockIdx.x;                       // 0 .. B*C-1
    const f4* xp = (const f4*)(x + (size_t)plane * (H_ * W_));
    f4 acc = (f4)(0.f);
    const int nq = (H_ * W_) / 4;                       // 36864 quads
    for (int i = threadIdx.x; i < nq; i += 256) {
        f4 v = __builtin_nontemporal_load(&xp[i]);
        acc += v;
    }
    red[threadIdx.x] = (acc.x + acc.y) + (acc.z + acc.w);
    __syncthreads();
    for (int off = 128; off > 0; off >>= 1) {
        if (threadIdx.x < off) red[threadIdx.x] += red[threadIdx.x + off];
        __syncthreads();
    }
    if (threadIdx.x == 0) pooled[plane] = red[0] * (1.0f / (H_ * W_));
}

// ---------------------------------------------------------------------------
// Kernel 2: kernel-generator MLP on WMMA.  Single workgroup, 8 waves.
//   L1: pooled[16x32] @ w1[32x128]  -> relu -> hdn (f16 in LDS)
//   L2: hdn[16x128]   @ w2[128x288] -> per-sample 3x3 depthwise kernels (f32)
// Fragment layouts per CDNA5 ISA 7.12.2 (wave32, 16-bit A/B, f32 C/D).
// ---------------------------------------------------------------------------
__global__ void mlp_wmma_kernel(const float* __restrict__ pooled,
                                const float* __restrict__ w1,
                                const float* __restrict__ b1,
                                const float* __restrict__ w2,
                                const float* __restrict__ b2,
                                float* __restrict__ kers) {
    __shared__ _Float16 hdn[16 * HID_];

    const int lane = threadIdx.x & 31;
    const int wv   = threadIdx.x >> 5;      // wave id 0..7
    const int g    = lane >> 4;             // half-wave group (0/1)
    const int n    = lane & 15;             // column within tile
    const int m    = lane & 15;             // A-matrix row

    // ---- Layer 1 ----
    // A fragment: 16-bit A 16x32 layout. elem i<8: K = 8g+i ; i>=8: K = 16+8g+(i-8)
    v16h a;
#pragma unroll
    for (int i = 0; i < 8; ++i) a[i]     = (_Float16)pooled[m * C_ + g * 8 + i];
#pragma unroll
    for (int i = 0; i < 8; ++i) a[8 + i] = (_Float16)pooled[m * C_ + 16 + g * 8 + i];

    // B fragment for this wave's N-tile: elem i<8: K = 8g+i ; i>=8: K = 16+8g+(i-8)
    const int n0 = wv * 16;
    v16h bfrag;
#pragma unroll
    for (int i = 0; i < 16; ++i)
        bfrag[i] = (_Float16)w1[(g * 8 + (i & 7) + 16 * (i >> 3)) * HID_ + n0 + n];

    v8f c = {};
    c = __builtin_amdgcn_wmma_f32_16x16x32_f16(false, a, false, bfrag,
                                               (short)0, c, false, false);

    // bias + relu -> f16 hidden activations in LDS
#pragma unroll
    for (int r = 0; r < 8; ++r) {
        const int mm = r + 8 * g;           // C/D layout: VGPR r -> M = r + 8*group
        float v = c[r] + b1[n0 + n];
        hdn[mm * HID_ + n0 + n] = (_Float16)(v > 0.f ? v : 0.f);
    }
    __syncthreads();

    // ---- Layer 2 ----  18 N-tiles of 16, K=128 in 4 chunks of 32
    for (int t = wv; t < 18; t += 8) {      // wave-uniform trip count
        const int tn0 = t * 16;
        v8f acc = {};
#pragma unroll
        for (int kc = 0; kc < HID_; kc += 32) {
            v16h a2, b2frag;
#pragma unroll
            for (int i = 0; i < 8; ++i) a2[i]     = hdn[m * HID_ + kc + g * 8 + i];
#pragma unroll
            for (int i = 0; i < 8; ++i) a2[8 + i] = hdn[m * HID_ + kc + 16 + g * 8 + i];
#pragma unroll
            for (int i = 0; i < 16; ++i)
                b2frag[i] = (_Float16)w2[(kc + g * 8 + (i & 7) + 16 * (i >> 3)) * KK_ + tn0 + n];
            acc = __builtin_amdgcn_wmma_f32_16x16x32_f16(false, a2, false, b2frag,
                                                         (short)0, acc, false, false);
        }
#pragma unroll
        for (int r = 0; r < 8; ++r) {
            const int mm = r + 8 * g;
            kers[mm * KK_ + tn0 + n] = acc[r] + b2[tn0 + n];
        }
    }
}

// ---------------------------------------------------------------------------
// Kernel 3: per-sample depthwise 3x3 conv, zero-padded.  Bandwidth bound:
// b128 in / b128 NT out per thread; x loads stay regular-temporal (each row
// feeds 3 output rows -> L0/L2 reuse keeps DRAM reads at ~1x).  The 9 weights
// are uniform per block (plane = blockIdx.y) so they lower to scalar loads.
// ---------------------------------------------------------------------------
__global__ void dconv3x3_kernel(const float* __restrict__ x,
                                const float* __restrict__ kers,
                                float* __restrict__ out) {
    const int plane = blockIdx.y;                            // b*NK + k == b*C + c
    const int pix   = blockIdx.x * blockDim.x + threadIdx.x; // quad index in plane
    const int wq    = pix % (W_ / 4);
    const int h     = pix / (W_ / 4);

    const float* kw = kers + plane * 9;                      // uniform -> s_load
    float k[9];
#pragma unroll
    for (int i = 0; i < 9; ++i) k[i] = kw[i];

    const float* xp = x + (size_t)plane * (H_ * W_);
    f4 acc = (f4)(0.f);
#pragma unroll
    for (int dr = -1; dr <= 1; ++dr) {
        const int r = h + dr;
        if (r < 0 || r >= H_) continue;
        const float* row = xp + r * W_ + wq * 4;
        const f4 cv = *(const f4*)row;
        const float lf = (wq > 0)          ? row[-1] : 0.f;
        const float rt = (wq < W_ / 4 - 1) ? row[4]  : 0.f;
        const float k0 = k[(dr + 1) * 3 + 0];
        const float k1 = k[(dr + 1) * 3 + 1];
        const float k2 = k[(dr + 1) * 3 + 2];
        acc.x += k0 * lf   + k1 * cv.x + k2 * cv.y;
        acc.y += k0 * cv.x + k1 * cv.y + k2 * cv.z;
        acc.z += k0 * cv.y + k1 * cv.z + k2 * cv.w;
        acc.w += k0 * cv.z + k1 * cv.w + k2 * rt;
    }
    // write-once output stream: non-temporal store, don't pollute L2
    __builtin_nontemporal_store(acc, (f4*)(out + (size_t)plane * (H_ * W_) + (size_t)pix * 4));
}

// ---------------------------------------------------------------------------
extern "C" void kernel_launch(void* const* d_in, const int* in_sizes, int n_in,
                              void* d_out, int out_size, void* d_ws, size_t ws_size,
                              hipStream_t stream) {
    (void)in_sizes; (void)n_in; (void)out_size; (void)ws_size;
    const float* x  = (const float*)d_in[0];
    const float* w1 = (const float*)d_in[1];
    const float* b1 = (const float*)d_in[2];
    const float* w2 = (const float*)d_in[3];
    const float* b2 = (const float*)d_in[4];
    float* out = (float*)d_out;

    float* pooled = (float*)d_ws;          // B*C   = 512  floats
    float* kers   = pooled + B_ * C_;      // B*KK  = 4608 floats

    pool_kernel<<<dim3(B_ * C_), dim3(256), 0, stream>>>(x, pooled);
    mlp_wmma_kernel<<<dim3(1), dim3(256), 0, stream>>>(pooled, w1, b1, w2, b2, kers);
    dconv3x3_kernel<<<dim3((H_ * W_ / 4) / 256, B_ * NK_), dim3(256), 0, stream>>>(x, kers, out);
}
